// PWBLinearLayer_42726334660724
// MI455X (gfx1250) — matrix operand order repaired and checked by reading, not compiled, not run on hardware
//
#include <hip/hip_runtime.h>

// ---------------------------------------------------------------------------
// PWB quantized linear: out = relu(x @ (k/127) + Q(b)), k = round(clip(W)*127)
// MI455X / gfx1250, wave32, v_wmma_f32_16x16x32_f16 with f32 accumulate.
// Weights quantized to exact integer k (f16) AND pre-transposed to [N][K] so
// the GEMM hot loop stages BOTH tiles with async global->LDS (ASYNCcnt) and
// does no transpose work. 1/127 folded into the epilogue.
// ---------------------------------------------------------------------------

typedef __attribute__((ext_vector_type(16))) _Float16 v16h;
typedef __attribute__((ext_vector_type(8)))  _Float16 v8h;
typedef __attribute__((ext_vector_type(8)))  float    v8f;
typedef __attribute__((ext_vector_type(4)))  int      v4i;

#define BLK_M 128
#define BLK_N 128
#define BK    32
#define LDSK  40   // padded K-stride (halfs): 80B row stride -> conflict-free b128

#if defined(__has_builtin)
#if __has_builtin(__builtin_amdgcn_global_load_async_to_lds_b128)
#define HAVE_ASYNC_LDS 1
#endif
#endif
#ifndef HAVE_ASYNC_LDS
#define HAVE_ASYNC_LDS 0
#endif

#if HAVE_ASYNC_LDS
// Builtin signature (from compiler diagnostic): param0 = v4i* in global AS,
// param1 = v4i* in LDS AS, then imm offset, imm cpol.
typedef __attribute__((address_space(1))) v4i* gvec_ptr;  // global v4i*
typedef __attribute__((address_space(3))) v4i* lvec_ptr;  // LDS v4i*
#endif

// ---- x: fp32 -> f16 (8 elements / thread, b128 loads, b128 store) ---------
__global__ __launch_bounds__(256) void cvt_x_kernel(
    const float* __restrict__ x, _Float16* __restrict__ x16, long long n) {
  long long i = ((long long)blockIdx.x * blockDim.x + threadIdx.x) * 8;
  if (i + 8 > n) return;
  float4 a = ((const float4*)(x + i))[0];
  float4 b = ((const float4*)(x + i))[1];
  v8h o;
  o[0] = (_Float16)a.x; o[1] = (_Float16)a.y;
  o[2] = (_Float16)a.z; o[3] = (_Float16)a.w;
  o[4] = (_Float16)b.x; o[5] = (_Float16)b.y;
  o[6] = (_Float16)b.z; o[7] = (_Float16)b.w;
  *(v8h*)(x16 + i) = o;
}

// ---- W[K][N] fp32 -> Wt[N][K] f16 holding k = round(clip(W)*127) ----------
// 64x64 LDS-tiled transpose: coalesced float4 reads, coalesced v8h writes.
#define TT   64
#define TLDS 72   // padded k-stride in halfs (144B rows, 16B aligned)
__global__ __launch_bounds__(256) void quant_wt_kernel(
    const float* __restrict__ w, _Float16* __restrict__ wt, int Kd, int Nd) {
  __shared__ _Float16 ldsT[TT * TLDS];  // [n][k] transposed tile
  const int k0 = blockIdx.y * TT;
  const int n1 = blockIdx.x * TT;
  const int t  = threadIdx.x;
#pragma unroll
  for (int i = 0; i < 4; ++i) {                 // 1024 float4 chunks / 256 thr
    int c    = t + i * 256;
    int row  = c >> 4;                          // k 0..63
    int col4 = (c & 15) * 4;                    // n 0..60
    float4 v = *(const float4*)(w + (size_t)(k0 + row) * Nd + n1 + col4);
    float vv[4] = {v.x, v.y, v.z, v.w};
#pragma unroll
    for (int j = 0; j < 4; ++j) {
      float q = fminf(fmaxf(vv[j], -1.f), 1.f) * 127.f;
      ldsT[(col4 + j) * TLDS + row] = (_Float16)rintf(q);  // RNE = jnp.round
    }
  }
  __syncthreads();
#pragma unroll
  for (int i = 0; i < 2; ++i) {                 // 512 v8h chunks / 256 thr
    int c    = t + i * 256;
    int rown = c >> 3;                          // n 0..63
    int col8 = (c & 7) * 8;                     // k 0..56
    v8h o = *(const v8h*)&ldsT[rown * TLDS + col8];
    *(v8h*)(wt + (size_t)(n1 + rown) * Kd + k0 + col8) = o;
  }
}

// ---- WMMA GEMM: out[M,N] = relu((x16 @ Wt^T)/127 + Q(bias)) ---------------
__global__ __launch_bounds__(256) void gemm_wmma_kernel(
    const _Float16* __restrict__ A,   // [M][K] f16 activations
    const _Float16* __restrict__ Bt,  // [N][K] f16 integer-valued weights^T
    const float* __restrict__ bias,   // [N] fp32 (quantized in epilogue)
    float* __restrict__ out,          // [M][N] fp32
    int M, int N, int K) {
  __shared__ _Float16 sA[2][BLK_M * LDSK];  // [M][K] tiles, padded
  __shared__ _Float16 sB[2][BLK_N * LDSK];  // [N][K] tiles, padded

  const int t  = threadIdx.x;
  const int m0 = blockIdx.y * BLK_M;
  const int n0 = blockIdx.x * BLK_N;

  const int lane = t & 31;
  const int wv   = t >> 5;          // 8 waves
  const int wm   = (wv >> 2) * 64;  // wave M base within tile (0/64)
  const int wn   = (wv & 3) * 32;   // wave N base within tile (0..96)
  const int lhi  = lane >> 4;       // half-wave select
  const int llo  = lane & 15;

  v8f acc[4][2];
#pragma unroll
  for (int mf = 0; mf < 4; ++mf)
#pragma unroll
    for (int nf = 0; nf < 2; ++nf) {
      v8f z = {0.f, 0.f, 0.f, 0.f, 0.f, 0.f, 0.f, 0.f};
      acc[mf][nf] = z;
    }

  // Stage a 128x32 A tile and a 128x32 B tile ([N][K], already transposed)
  // into LDS. Identical layouts -> both go through the async path.
  auto stage = [&](int buf, int k0) {
#pragma unroll
    for (int i = 0; i < 2; ++i) {             // 512 b128 chunks each / 256 thr
      int c    = t + i * 256;
      int row  = c >> 2;                      // 0..127
      int col8 = (c & 3) * 8;                 // 0,8,16,24
      const _Float16* gpA = A  + (size_t)(m0 + row) * K + k0 + col8;
      const _Float16* gpB = Bt + (size_t)(n0 + row) * K + k0 + col8;
      _Float16* lpA = &sA[buf][row * LDSK + col8];
      _Float16* lpB = &sB[buf][row * LDSK + col8];
#if HAVE_ASYNC_LDS
      // gfx1250: GLOBAL_LOAD_ASYNC_TO_LDS_B128, tracked with ASYNCcnt.
      __builtin_amdgcn_global_load_async_to_lds_b128((gvec_ptr)gpA,
                                                     (lvec_ptr)lpA, 0, 0);
      __builtin_amdgcn_global_load_async_to_lds_b128((gvec_ptr)gpB,
                                                     (lvec_ptr)lpB, 0, 0);
#else
      *(v8h*)lpA = *(const v8h*)gpA;
      *(v8h*)lpB = *(const v8h*)gpB;
#endif
    }
  };

  const int NK = K / BK;
  stage(0, 0);
#if HAVE_ASYNC_LDS
  asm volatile("s_wait_asynccnt 0x0" ::: "memory");
#endif
  __syncthreads();

  for (int kt = 0; kt < NK; ++kt) {
    const int buf = kt & 1;
    if (kt + 1 < NK) stage(buf ^ 1, (kt + 1) * BK);  // prefetch next tile

    // A fragments: 16x32 f16 layout — lanes 0-15 hold K 0-7 & 16-23,
    // lanes 16-31 hold K 8-15 & 24-31 (two b128 LDS reads each).
    v16h af[4];
#pragma unroll
    for (int mf = 0; mf < 4; ++mf) {
      const _Float16* p = &sA[buf][(wm + mf * 16 + llo) * LDSK];
      v8h lo = *(const v8h*)(p + lhi * 8);
      v8h hi = *(const v8h*)(p + 16 + lhi * 8);
#pragma unroll
      for (int j = 0; j < 8; ++j) { af[mf][j] = lo[j]; af[mf][8 + j] = hi[j]; }
    }
    // B fragments: 32x16 f16 layout — lanes 0-15 hold K 0-7 & 16-23,
    // lanes 16-31 hold K 8-15 & 24-31 (same K order as A).
    v16h bf[2];
#pragma unroll
    for (int nf = 0; nf < 2; ++nf) {
      const _Float16* p = &sB[buf][(wn + nf * 16 + llo) * LDSK + lhi * 8];
      v8h lo = *(const v8h*)p;
      v8h hi = *(const v8h*)(p + 16);
#pragma unroll
      for (int j = 0; j < 8; ++j) { bf[nf][j] = lo[j]; bf[nf][8 + j] = hi[j]; }
    }

#pragma unroll
    for (int mf = 0; mf < 4; ++mf)
#pragma unroll
      for (int nf = 0; nf < 2; ++nf)
        acc[mf][nf] = __builtin_amdgcn_wmma_f32_16x16x32_f16(
            false, af[mf], false, bf[nf], (short)0, acc[mf][nf], false, false);

#if HAVE_ASYNC_LDS
    asm volatile("s_wait_asynccnt 0x0" ::: "memory");
#endif
    __syncthreads();
  }

  // Epilogue: scale by 1/127, add quantized bias, relu, coalesced f32 store.
  const float inv127 = 1.0f / 127.0f;
#pragma unroll
  for (int nf = 0; nf < 2; ++nf) {
    const int n = n0 + wn + nf * 16 + llo;
    float bq = fminf(fmaxf(bias[n], -1.f), 1.f);
    bq = rintf(bq * 127.f) * inv127;
#pragma unroll
    for (int mf = 0; mf < 4; ++mf) {
#pragma unroll
      for (int r = 0; r < 8; ++r) {
        const int m = m0 + wm + mf * 16 + lhi * 8 + r;
        float v = acc[mf][nf][r] * inv127 + bq;
        out[(size_t)m * N + n] = fmaxf(v, 0.f);
      }
    }
  }
}

extern "C" void kernel_launch(void* const* d_in, const int* in_sizes, int n_in,
                              void* d_out, int out_size, void* d_ws, size_t ws_size,
                              hipStream_t stream) {
  const float* x    = (const float*)d_in[0];
  const float* w    = (const float*)d_in[1];
  const float* bias = (const float*)d_in[2];
  float* out = (float*)d_out;

  const long long Dout = in_sizes[2];
  const long long Din  = in_sizes[1] / Dout;
  const long long Bm   = (long long)in_sizes[0] / Din;

  _Float16* x16 = (_Float16*)d_ws;                 // [Bm][Din]
  _Float16* wt16 = x16 + (size_t)Bm * Din;         // [Dout][Din] (transposed)

  const long long nx = Bm * Din;

  cvt_x_kernel<<<(unsigned)((nx / 8 + 255) / 256), 256, 0, stream>>>(x, x16, nx);

  dim3 tgrid((unsigned)(Dout / TT), (unsigned)(Din / TT));
  quant_wt_kernel<<<tgrid, 256, 0, stream>>>(w, wt16, (int)Din, (int)Dout);

  dim3 grid((unsigned)(Dout / BLK_N), (unsigned)(Bm / BLK_M));
  gemm_wmma_kernel<<<grid, 256, 0, stream>>>(x16, wt16, bias, out,
                                             (int)Bm, (int)Dout, (int)Din);
}
